// DGCNN_param_57904749085240
// MI455X (gfx1250) — compile-verified, matrix operand order checked by reading, not run on hardware
//
#include <hip/hip_runtime.h>

// ---------------------------------------------------------------------------
// DGCNN forward for MI455X (gfx1250, wave32). All dense contractions use
// v_wmma_f32_16x16x32_f16 (f16 in, f32 accumulate). Edge features are built
// in LDS in 64-row K-chunks; BN + LeakyReLU + max-over-k fused in epilogues.
// ---------------------------------------------------------------------------

typedef __attribute__((ext_vector_type(16))) _Float16 v16h;
typedef __attribute__((ext_vector_type(8)))  float    v8f;

#define BATCH 16
#define NPTS  1024
#define KNN_K 20
#define EC_COLS 160   // 8 points * 20 neighbors per block
#define EC_KC   64    // K-chunk rows staged in LDS

// WMMA 16x16x32 f16 fragment load (row-major f16 matrix; for B-operand,
// "row" is the column index of the K x 16 tile). Layout per ISA 7.12.2:
// lane half h=lane>>4; 32-bit pair v: K = (v<4) ? 8h+2v : 16+8h+2(v-4).
__device__ __forceinline__ v16h load_frag(const _Float16* base, int stride,
                                          int row, int kbase, int lane) {
  union { unsigned u[8]; v16h h; } r;
  const _Float16* p = base + (size_t)row * stride + kbase;
  const int hh = (lane >> 4) & 1;
#pragma unroll
  for (int vv = 0; vv < 8; ++vv) {
    int k = (vv < 4) ? (8 * hh + 2 * vv) : (16 + 8 * hh + 2 * (vv - 4));
    r.u[vv] = *(const unsigned*)(p + k);
  }
  return r.h;
}

__device__ __forceinline__ v8f wmma_f16(v16h a, v16h b, v8f c) {
  return __builtin_amdgcn_wmma_f32_16x16x32_f16(false, a, false, b,
                                                (short)0, c, false, false);
}

__device__ __forceinline__ void atomicMaxF(float* addr, float val) {
  int* ai = reinterpret_cast<int*>(addr);
  int old = __float_as_int(*addr);
  while (__int_as_float(old) < val) {
    int assumed = old;
    old = atomicCAS(ai, assumed, __float_as_int(val));
    if (old == assumed) break;
  }
}

// ---------------- x transpose: (B,N,3) -> (B,3,N) ----------------
__global__ void transpose_x(const float* __restrict__ x, float* __restrict__ xT) {
  int i = blockIdx.x * 256 + threadIdx.x;
  if (i >= BATCH * NPTS * 3) return;
  int b = i / (NPTS * 3), rem = i % (NPTS * 3);
  int n = rem / 3, c = rem % 3;
  xT[(size_t)b * 3 * NPTS + (size_t)c * NPTS + n] = x[i];
}

// ---------------- f32 -> f16 with zero padding ----------------
__global__ void convert_pad(const float* __restrict__ src, _Float16* __restrict__ dst,
                            int R, int C, int Rpad, int Cpad) {
  int i = blockIdx.x * 256 + threadIdx.x;
  if (i >= Rpad * Cpad) return;
  int r = i / Cpad, c = i % Cpad;
  dst[i] = (r < R && c < C) ? (_Float16)src[(size_t)r * C + c] : (_Float16)0.f;
}

// ---------------- pooling init ----------------
__global__ void init_pool(float* __restrict__ hmax, float* __restrict__ hsum) {
  int i = blockIdx.x * 256 + threadIdx.x;
  if (i >= BATCH * 1024) return;
  hmax[i] = -3.4e38f;
  hsum[i] = 0.f;
}

// ---------------- KNN: top-20 by pd = -||xi-xj||^2 (self included) ---------
template <int C>
__global__ void knn_kernel(const float* __restrict__ x, int bstride,
                           int* __restrict__ idx) {
  const int b = blockIdx.y;
  const int i = blockIdx.x * 128 + threadIdx.x;
  const float* xb = x + (size_t)b * bstride;
  float xi[C];
#pragma unroll
  for (int c = 0; c < C; ++c) xi[c] = xb[(size_t)c * NPTS + i];
  float best[KNN_K];
  int besti[KNN_K];
#pragma unroll
  for (int k = 0; k < KNN_K; ++k) { best[k] = -3.4e38f; besti[k] = 0; }

  __shared__ float xs[C][64];
  for (int jt = 0; jt < NPTS; jt += 64) {
    __syncthreads();
    for (int e = threadIdx.x; e < C * 64; e += 128) {
      int c = e >> 6, t = e & 63;
      xs[c][t] = xb[(size_t)c * NPTS + jt + t];
    }
    __syncthreads();
    for (int t = 0; t < 64; ++t) {
      float d2 = 0.f;
#pragma unroll
      for (int c = 0; c < C; ++c) { float d = xi[c] - xs[c][t]; d2 += d * d; }
      float sc = -d2;
      int j = jt + t;
#pragma unroll
      for (int k = 0; k < KNN_K; ++k) {       // register insertion sort
        if (sc > best[k]) {
          float tv = best[k]; best[k] = sc; sc = tv;
          int ti = besti[k]; besti[k] = j; j = ti;
        }
      }
    }
  }
  int* op = idx + ((size_t)b * NPTS + i) * KNN_K;
#pragma unroll
  for (int k = 0; k < KNN_K; ++k) op[k] = besti[k];
}

// ---------------- EdgeConv: WMMA GEMM + BN + LReLU + max over k ------------
// Block: 16 output channels x 8 points (160 gathered columns), 5 waves.
__global__ void edgeconv_kernel(const float* __restrict__ x, int bstride, int C,
                                const int* __restrict__ idx,
                                const _Float16* __restrict__ wf16, // O x 256 padded
                                const float* __restrict__ bias,
                                const float* __restrict__ g,
                                const float* __restrict__ be,
                                const float* __restrict__ m,
                                const float* __restrict__ v,
                                float* __restrict__ out, int obstride) {
  const int ob = blockIdx.x * 16;
  const int nb = blockIdx.y * 8;
  const int b  = blockIdx.z;
  const int tid  = threadIdx.x;      // 0..159
  const int wave = tid >> 5;         // 0..4
  const int lane = tid & 31;

  __shared__ _Float16 Bm[EC_COLS][EC_KC];   // 20 KB, [column][k-in-chunk]
  __shared__ float    yt[16][EC_COLS];      // 10 KB
  __shared__ int      nidx[EC_COLS];
  __shared__ float    sA[16], sB[16];

  const float* xb = x + (size_t)b * bstride;

  if (tid < EC_COLS) {
    int p = tid / KNN_K, k = tid % KNN_K;
    nidx[tid] = idx[((size_t)b * NPTS + nb + p) * KNN_K + k];
  }
  if (tid < 16) {
    int o = ob + tid;
    float s = g[o] * rsqrtf(v[o] + 1e-5f);
    sA[tid] = s;
    sB[tid] = (bias[o] - m[o]) * s + be[o];
  }

  const int K2 = 2 * C;
  const int K2pad = (K2 + 31) & ~31;
  const int t0 = 2 * wave, t1 = 2 * wave + 1;
  v8f acc0 = {}; v8f acc1 = {};

  for (int kc = 0; kc < K2pad; kc += EC_KC) {
    __syncthreads();
    for (int e = tid; e < EC_COLS * EC_KC; e += EC_COLS) {   // uniform trips
      int col = e >> 6, r = e & 63;
      int rr = kc + r;
      float val = 0.f;
      int p = col / KNN_K;
      if (rr < C)        val = xb[(size_t)rr * NPTS + nidx[col]] -
                               xb[(size_t)rr * NPTS + nb + p];
      else if (rr < K2)  val = xb[(size_t)(rr - C) * NPTS + nb + p];
      Bm[col][r] = (_Float16)val;
    }
    __syncthreads();
    int rows = K2pad - kc; if (rows > EC_KC) rows = EC_KC;
    int steps = rows / 32;
    for (int s = 0; s < steps; ++s) {
      v16h a  = load_frag(wf16 + (size_t)ob * 256, 256, lane & 15, kc + s * 32, lane);
      v16h b0 = load_frag(&Bm[t0 * 16][0], EC_KC, lane & 15, s * 32, lane);
      acc0 = wmma_f16(a, b0, acc0);
      v16h b1 = load_frag(&Bm[t1 * 16][0], EC_KC, lane & 15, s * 32, lane);
      acc1 = wmma_f16(a, b1, acc1);
    }
  }

  const int hh = lane >> 4, nl = lane & 15;
#pragma unroll
  for (int r = 0; r < 8; ++r) {
    int M = 8 * hh + r;
    float v0 = acc0[r] * sA[M] + sB[M];
    float v1 = acc1[r] * sA[M] + sB[M];
    yt[M][t0 * 16 + nl] = v0 > 0.f ? v0 : 0.2f * v0;
    yt[M][t1 * 16 + nl] = v1 > 0.f ? v1 : 0.2f * v1;
  }
  __syncthreads();
  if (tid < 128) {
    int o = tid & 15, p = tid >> 4;
    float mx = -3.4e38f;
#pragma unroll
    for (int k = 0; k < KNN_K; ++k) mx = fmaxf(mx, yt[o][p * KNN_K + k]);
    out[(size_t)b * obstride + (size_t)(ob + o) * NPTS + nb + p] = mx;
  }
}

// ---------------- lc conv 512->1024 + BN + LReLU + fused max/sum pool ------
__global__ void lcconv_kernel(const float* __restrict__ x,  // (B,512,N) concat
                              const _Float16* __restrict__ wf16, // 1024 x 512
                              const float* __restrict__ bias,
                              const float* __restrict__ g,
                              const float* __restrict__ be,
                              const float* __restrict__ m,
                              const float* __restrict__ v,
                              float* __restrict__ hmax, float* __restrict__ hsum) {
  const int ob = blockIdx.x * 16;
  const int n0 = blockIdx.y * 64;
  const int b  = blockIdx.z;
  const int tid  = threadIdx.x;     // 0..127
  const int wave = tid >> 5;        // 0..3, one 16-col tile each
  const int lane = tid & 31;

  __shared__ _Float16 Bm[64][64];   // 8 KB
  __shared__ float    yt[16][64];
  __shared__ float    sA[16], sB[16];

  const float* xb = x + (size_t)b * 512 * NPTS;
  if (tid < 16) {
    int o = ob + tid;
    float s = g[o] * rsqrtf(v[o] + 1e-5f);
    sA[tid] = s;
    sB[tid] = (bias[o] - m[o]) * s + be[o];
  }
  v8f acc = {};
  for (int kc = 0; kc < 512; kc += 64) {
    __syncthreads();
    for (int e = tid; e < 64 * 64; e += 128) {   // uniform trips
      int col = e >> 6, r = e & 63;
      Bm[col][r] = (_Float16)xb[(size_t)(kc + r) * NPTS + n0 + col];
    }
    __syncthreads();
    for (int s = 0; s < 2; ++s) {
      v16h a  = load_frag(wf16 + (size_t)ob * 512, 512, lane & 15, kc + s * 32, lane);
      v16h bb = load_frag(&Bm[wave * 16][0], 64, lane & 15, s * 32, lane);
      acc = wmma_f16(a, bb, acc);
    }
  }
  const int hh = lane >> 4, nl = lane & 15;
#pragma unroll
  for (int r = 0; r < 8; ++r) {
    int M = 8 * hh + r;
    float val = acc[r] * sA[M] + sB[M];
    yt[M][wave * 16 + nl] = val > 0.f ? val : 0.2f * val;
  }
  __syncthreads();
  if (tid < 16) {
    float mx = -3.4e38f, sm = 0.f;
    for (int col = 0; col < 64; ++col) { float vv = yt[tid][col]; mx = fmaxf(mx, vv); sm += vv; }
    atomicMaxF(&hmax[(size_t)b * 1024 + ob + tid], mx);
    atomicAdd(&hsum[(size_t)b * 1024 + ob + tid], sm);
  }
}

// ---------------- build h = [max, mean] as f16 (16 x 2048) ----------------
__global__ void build_h(const float* __restrict__ hmax, const float* __restrict__ hsum,
                        _Float16* __restrict__ hf16) {
  int i = blockIdx.x * 256 + threadIdx.x;
  if (i >= BATCH * 2048) return;
  int b = i / 2048, o = i % 2048;
  float val = (o < 1024) ? hmax[(size_t)b * 1024 + o]
                         : hsum[(size_t)b * 1024 + (o - 1024)] * (1.f / NPTS);
  hf16[i] = (_Float16)val;
}

// ---------------- FC layer: out(16,O) = act(bn(A(16,K) @ W^T)) -------------
__global__ void fc_kernel(const _Float16* __restrict__ A, int K,
                          const _Float16* __restrict__ W,  // Opad rows, stride K
                          const float* __restrict__ bias,
                          const float* __restrict__ g, const float* __restrict__ be,
                          const float* __restrict__ m, const float* __restrict__ v,
                          int useBN, int useAct,
                          float* __restrict__ outF32, _Float16* __restrict__ outF16,
                          int O) {
  const int ot = blockIdx.x * 16;
  const int lane = threadIdx.x & 31;
  v8f acc = {};
  for (int kc = 0; kc < K; kc += 32) {
    v16h a  = load_frag(A, K, lane & 15, kc, lane);                 // row = batch
    v16h bb = load_frag(W + (size_t)ot * K, K, lane & 15, kc, lane); // row = o
    acc = wmma_f16(a, bb, acc);
  }
  const int hh = lane >> 4;
  const int o = ot + (lane & 15);
  float sAv = 1.f, sBv = 0.f;
  if (o < O) {
    float bz = bias ? bias[o] : 0.f;
    if (useBN) {
      float s = g[o] * rsqrtf(v[o] + 1e-5f);
      sAv = s; sBv = (bz - m[o]) * s + be[o];
    } else sBv = bz;
  }
#pragma unroll
  for (int r = 0; r < 8; ++r) {
    int M = 8 * hh + r;   // batch row
    float val = acc[r] * sAv + sBv;
    if (useAct) val = val > 0.f ? val : 0.2f * val;
    if (o < O) {
      outF32[(size_t)M * O + o] = val;
      if (outF16) outF16[(size_t)M * O + o] = (_Float16)val;
    }
  }
}

// ---------------------------------------------------------------------------
extern "C" void kernel_launch(void* const* d_in, const int* in_sizes, int n_in,
                              void* d_out, int out_size, void* d_ws, size_t ws_size,
                              hipStream_t stream) {
  (void)in_sizes; (void)n_in; (void)out_size; (void)ws_size;
  const float* x = (const float*)d_in[0];
  const float *cw[4], *cb[4], *cg[4], *cbe[4], *cm[4], *cv[4];
  for (int i = 0; i < 4; ++i) {
    cw[i]  = (const float*)d_in[1 + 6 * i + 0];
    cb[i]  = (const float*)d_in[1 + 6 * i + 1];
    cg[i]  = (const float*)d_in[1 + 6 * i + 2];
    cbe[i] = (const float*)d_in[1 + 6 * i + 3];
    cm[i]  = (const float*)d_in[1 + 6 * i + 4];
    cv[i]  = (const float*)d_in[1 + 6 * i + 5];
  }
  const float* lc_w = (const float*)d_in[25]; const float* lc_b = (const float*)d_in[26];
  const float* lc_g = (const float*)d_in[27]; const float* lc_be = (const float*)d_in[28];
  const float* lc_m = (const float*)d_in[29]; const float* lc_v = (const float*)d_in[30];
  const float* l0_w = (const float*)d_in[31]; const float* l0_g = (const float*)d_in[32];
  const float* l0_be = (const float*)d_in[33]; const float* l0_m = (const float*)d_in[34];
  const float* l0_v = (const float*)d_in[35];
  const float* l1_w = (const float*)d_in[36]; const float* l1_b = (const float*)d_in[37];
  const float* l1_g = (const float*)d_in[38]; const float* l1_be = (const float*)d_in[39];
  const float* l1_m = (const float*)d_in[40]; const float* l1_v = (const float*)d_in[41];
  const float* ow = (const float*)d_in[42]; const float* ob = (const float*)d_in[43];

  // workspace carve-out
  char* w = (char*)d_ws; size_t off = 0;
  auto alloc = [&](size_t bytes) -> void* {
    void* p = w + off; off = (off + bytes + 255) & ~(size_t)255; return p;
  };
  float*     xT     = (float*)alloc((size_t)BATCH * 3 * NPTS * 4);
  float*     concat = (float*)alloc((size_t)BATCH * 512 * NPTS * 4);
  int*       idx    = (int*)alloc((size_t)BATCH * NPTS * KNN_K * 4);
  const int  Oc[4] = {64, 64, 128, 256};
  _Float16*  wcf[4];
  for (int i = 0; i < 4; ++i) wcf[i] = (_Float16*)alloc((size_t)Oc[i] * 256 * 2);
  _Float16*  lcwf = (_Float16*)alloc((size_t)1024 * 512 * 2);
  _Float16*  l0wf = (_Float16*)alloc((size_t)512 * 2048 * 2);
  _Float16*  l1wf = (_Float16*)alloc((size_t)256 * 512 * 2);
  _Float16*  owf  = (_Float16*)alloc((size_t)48 * 256 * 2);
  float*     hmax = (float*)alloc((size_t)BATCH * 1024 * 4);
  float*     hsum = (float*)alloc((size_t)BATCH * 1024 * 4);
  _Float16*  hf16 = (_Float16*)alloc((size_t)BATCH * 2048 * 2);
  float*     h0f  = (float*)alloc((size_t)BATCH * 512 * 4);
  _Float16*  h0h  = (_Float16*)alloc((size_t)BATCH * 512 * 2);
  float*     h1f  = (float*)alloc((size_t)BATCH * 256 * 4);
  _Float16*  h1h  = (_Float16*)alloc((size_t)BATCH * 256 * 2);

  auto cdiv = [](int a, int b) { return (a + b - 1) / b; };

  // x transpose + weight conversions
  transpose_x<<<cdiv(BATCH * NPTS * 3, 256), 256, 0, stream>>>(x, xT);
  const int Ci2[4] = {6, 128, 128, 256};
  for (int i = 0; i < 4; ++i)
    convert_pad<<<cdiv(Oc[i] * 256, 256), 256, 0, stream>>>(cw[i], wcf[i], Oc[i], Ci2[i], Oc[i], 256);
  convert_pad<<<cdiv(1024 * 512, 256), 256, 0, stream>>>(lc_w, lcwf, 1024, 512, 1024, 512);
  convert_pad<<<cdiv(512 * 2048, 256), 256, 0, stream>>>(l0_w, l0wf, 512, 2048, 512, 2048);
  convert_pad<<<cdiv(256 * 512, 256), 256, 0, stream>>>(l1_w, l1wf, 256, 512, 256, 512);
  convert_pad<<<cdiv(48 * 256, 256), 256, 0, stream>>>(ow, owf, 40, 256, 48, 256);
  init_pool<<<cdiv(BATCH * 1024, 256), 256, 0, stream>>>(hmax, hsum);

  // EdgeConv stack; each layer's output is a channel-slice of `concat`
  const int   chOff[4] = {0, 64, 128, 256};
  const float* layerIn[4] = {xT, concat + 0 * NPTS, concat + 64 * NPTS, concat + 128 * NPTS};
  const int   inBS[4] = {3 * NPTS, 512 * NPTS, 512 * NPTS, 512 * NPTS};
  const int   Cin[4] = {3, 64, 64, 128};
  for (int L = 0; L < 4; ++L) {
    dim3 kg(NPTS / 128, BATCH);
    if (Cin[L] == 3)        knn_kernel<3><<<kg, 128, 0, stream>>>(layerIn[L], inBS[L], idx);
    else if (Cin[L] == 64)  knn_kernel<64><<<kg, 128, 0, stream>>>(layerIn[L], inBS[L], idx);
    else                    knn_kernel<128><<<kg, 128, 0, stream>>>(layerIn[L], inBS[L], idx);
    dim3 eg(Oc[L] / 16, NPTS / 8, BATCH);
    edgeconv_kernel<<<eg, EC_COLS, 0, stream>>>(
        layerIn[L], inBS[L], Cin[L], idx, wcf[L],
        cb[L], cg[L], cbe[L], cm[L], cv[L],
        concat + (size_t)chOff[L] * NPTS, 512 * NPTS);
  }

  // lc conv + fused global max/sum pooling
  dim3 lg(1024 / 16, NPTS / 64, BATCH);
  lcconv_kernel<<<lg, 128, 0, stream>>>(concat, lcwf, lc_b, lc_g, lc_be, lc_m, lc_v,
                                        hmax, hsum);
  build_h<<<cdiv(BATCH * 2048, 256), 256, 0, stream>>>(hmax, hsum, hf16);

  // FC head (M=16 batch rows fit one WMMA tile)
  fc_kernel<<<512 / 16, 32, 0, stream>>>(hf16, 2048, l0wf, nullptr,
                                         l0_g, l0_be, l0_m, l0_v, 1, 1, h0f, h0h, 512);
  fc_kernel<<<256 / 16, 32, 0, stream>>>(h0h, 512, l1wf, l1_b,
                                         l1_g, l1_be, l1_m, l1_v, 1, 1, h1f, h1h, 256);
  fc_kernel<<<48 / 16, 32, 0, stream>>>(h1h, 256, owf, ob,
                                        nullptr, nullptr, nullptr, nullptr, 0, 0,
                                        (float*)d_out, nullptr, 40);
}
